// MultiHeadAttention_12773232738643
// MI455X (gfx1250) — matrix operand order repaired and checked by reading, not compiled, not run on hardware
//
#include <hip/hip_runtime.h>
#include <hip/hip_bf16.h>

// ---------------------------------------------------------------------------
// GQA attention block for MI455X (gfx1250, wave32, WMMA).
//   q = rope(x@Wq), k = rope(x@Wk), v = x@Wv
//   out = softmax(q k^T / sqrt(d) + alibi + causal) v @ Wo
// All matmuls via v_wmma_f32_16x16x32_f16 (f16 in, fp32 accumulate).
// Flash-style attention: scores never touch HBM (1.1 GB avoided).
// GEMM tiles move global->LDS via CDNA5 async loads (ASYNCcnt), double-buffered.
// ---------------------------------------------------------------------------

#define D_MODEL 2048
#define N_HEADS 16
#define N_KV    4
#define HD      128
#define B_SZ    4
#define T_SZ    2048
#define M_ROWS  (B_SZ * T_SZ)   // 8192

typedef _Float16 half8 __attribute__((ext_vector_type(8)));
typedef _Float16 v16h  __attribute__((ext_vector_type(16)));
typedef float    v8f   __attribute__((ext_vector_type(8)));

__device__ __forceinline__ v16h pack16(half8 lo, half8 hi) {
  v16h r;
#pragma unroll
  for (int i = 0; i < 8; i++) { r[i] = lo[i]; r[i + 8] = hi[i]; }
  return r;
}

// A-operand fragment (16x32 f16, MxK):
//   lane<16 : row M = lane,    K = {0..7, 16..23}
//   lane>=16: row M = lane-16, K = {8..15, 24..31}
__device__ __forceinline__ v16h ldfragA(const _Float16* rowbase, int rowStride, int lane) {
  const _Float16* p = rowbase + (size_t)(lane & 15) * rowStride + ((lane < 16) ? 0 : 8);
  half8 lo = *(const half8*)p;          // K block +0..7
  half8 hi = *(const half8*)(p + 16);   // K block +16..23
  return pack16(lo, hi);
}

// B-operand fragment (32x16 f16, KxN) read from N-major (transposed) storage:
//   lane<16 : column N = lane,    K = {0..7, 16..23}
//   lane>=16: column N = lane-16, K = {8..15, 24..31}
__device__ __forceinline__ v16h ldfragB(const _Float16* rowbase, int rowStride, int lane) {
  const _Float16* p = rowbase + (size_t)(lane & 15) * rowStride + ((lane < 16) ? 0 : 8);
  half8 lo = *(const half8*)p;
  half8 hi = *(const half8*)(p + 16);
  return pack16(lo, hi);
}

__device__ __forceinline__ v8f wmma32(v16h a, v16h b, v8f c) {
  return __builtin_amdgcn_wmma_f32_16x16x32_f16(false, a, false, b, (short)0, c, false, false);
}

// CDNA5 async global->LDS copy (16B per lane), tracked by ASYNCcnt.
// LDS offset = low 32 bits of the generic pointer (addrspacecast(3->flat)
// keeps the workgroup-relative LDS offset in the low dword).
__device__ __forceinline__ void async_ld_b128(void* lds_ptr, const void* gptr) {
  unsigned int loff = (unsigned int)(uintptr_t)lds_ptr;
  asm volatile("global_load_async_to_lds_b128 %0, %1, off"
               :
               : "v"(loff), "v"(gptr)
               : "memory");
}
__device__ __forceinline__ void wait_async0() {
  asm volatile("s_wait_asynccnt 0x0" ::: "memory");
}

// ---------------------------------------------------------------------------
// C[M,N] = A[M,K] @ B[K,N], A row-major f16, Bt = B^T (N x K) row-major f16,
// C fp32 row-major.  Workgroup = 8 waves, 128x128 macro-tile.
// Double-buffered LDS tiles filled by async loads; one barrier per k-step.
// Each wave: 2x4 grid of 16x16 WMMA tiles (32 rows x 64 cols).
// ---------------------------------------------------------------------------
__global__ __launch_bounds__(256)
void gemm_f16t(const _Float16* __restrict__ A, const _Float16* __restrict__ Bt,
               float* __restrict__ C, int M, int N, int K) {
  __shared__ __align__(16) _Float16 As[2][128 * 32];
  __shared__ __align__(16) _Float16 Bs[2][128 * 32];

  const int t    = threadIdx.x;
  const int lane = t & 31;
  const int wid  = t >> 5;
  const int wm   = (wid & 3) * 32;   // wave row base inside macro-tile
  const int wn   = (wid >> 2) * 64;  // wave col base inside macro-tile
  const int m0   = blockIdx.y * 128;
  const int n0   = blockIdx.x * 128;

  // this thread's two 8-half chunks of the 128x32 tile
  const int c0 = t, c1 = t + 256;
  const int rowc0 = c0 >> 2, kc0 = (c0 & 3) * 8;
  const int rowc1 = c1 >> 2, kc1 = (c1 & 3) * 8;

  auto issue_tile = [&](int k0, int buf) {
    async_ld_b128(&As[buf][rowc0 * 32 + kc0], &A [(size_t)(m0 + rowc0) * K + k0 + kc0]);
    async_ld_b128(&Bs[buf][rowc0 * 32 + kc0], &Bt[(size_t)(n0 + rowc0) * K + k0 + kc0]);
    async_ld_b128(&As[buf][rowc1 * 32 + kc1], &A [(size_t)(m0 + rowc1) * K + k0 + kc1]);
    async_ld_b128(&Bs[buf][rowc1 * 32 + kc1], &Bt[(size_t)(n0 + rowc1) * K + k0 + kc1]);
  };

  v8f acc[2][4];
#pragma unroll
  for (int mt = 0; mt < 2; mt++)
#pragma unroll
    for (int nt = 0; nt < 4; nt++) acc[mt][nt] = 0.0f;

  issue_tile(0, 0);
  const int nk = K >> 5;
  for (int kk = 0; kk < nk; kk++) {
    const int buf = kk & 1;
    wait_async0();        // my async writes for this buffer are done
    __syncthreads();      // everyone's writes done; everyone done reading other buffer
    if (kk + 1 < nk) issue_tile((kk + 1) << 5, buf ^ 1);

    v16h af[2], bf[4];
#pragma unroll
    for (int mt = 0; mt < 2; mt++) af[mt] = ldfragA(&As[buf][(wm + mt * 16) * 32], 32, lane);
#pragma unroll
    for (int nt = 0; nt < 4; nt++) bf[nt] = ldfragB(&Bs[buf][(wn + nt * 16) * 32], 32, lane);
#pragma unroll
    for (int mt = 0; mt < 2; mt++)
#pragma unroll
      for (int nt = 0; nt < 4; nt++) acc[mt][nt] = wmma32(af[mt], bf[nt], acc[mt][nt]);
  }

  // C/D layout: lanes 0-15 -> col=lane, rows 0-7 ; lanes 16-31 -> col=lane-16, rows 8-15
  const int r  = lane & 15;
  const int hs = lane >> 4;
#pragma unroll
  for (int mt = 0; mt < 2; mt++)
#pragma unroll
    for (int nt = 0; nt < 4; nt++) {
      int col = n0 + wn + nt * 16 + r;
#pragma unroll
      for (int i = 0; i < 8; i++) {
        int row = m0 + wm + mt * 16 + hs * 8 + i;
        C[(size_t)row * N + col] = acc[mt][nt][i];
      }
    }
}

// ---------------------------------------------------------------------------
// Elementwise helpers
// ---------------------------------------------------------------------------
__global__ void f32_to_f16(const float* __restrict__ in, _Float16* __restrict__ out, int n) {
  int i = blockIdx.x * blockDim.x + threadIdx.x;
  if (i < n) out[i] = (_Float16)in[i];
}

// in: R x Ccols fp32 row-major  ->  out: Ccols x R f16 row-major
__global__ void transpose_to_f16(const float* __restrict__ in, _Float16* __restrict__ out,
                                 int R, int Ccols) {
  int idx = blockIdx.x * blockDim.x + threadIdx.x;
  if (idx >= R * Ccols) return;
  int cc = idx % Ccols;
  int rr = idx / Ccols;
  out[(size_t)cc * R + rr] = (_Float16)in[(size_t)rr * Ccols + cc];
}

// RoPE + repack: Xf (B*T, NH*HD) fp32  ->  Xh (B, NH, T, HD) f16
__global__ void rope_pack(const float* __restrict__ Xf, _Float16* __restrict__ Xh, int NH) {
  int idx = blockIdx.x * blockDim.x + threadIdx.x;
  int total = B_SZ * NH * T_SZ * 64;
  if (idx >= total) return;
  int d  = idx & 63;
  int r1 = idx >> 6;
  int tt = r1 % T_SZ;
  int r2 = r1 / T_SZ;
  int h  = r2 % NH;
  int b  = r2 / NH;
  const float* row = Xf + (size_t)(b * T_SZ + tt) * (NH * HD) + h * HD;
  float x1 = row[d], x2 = row[d + 64];
  float theta = __powf(10000.0f, -(float)d * (1.0f / 64.0f));
  float s, c;
  __sincosf((float)tt * theta, &s, &c);
  _Float16* orow = Xh + (size_t)((b * NH + h) * T_SZ + tt) * HD;
  orow[d]      = (_Float16)(x1 * c - x2 * s);
  orow[d + 64] = (_Float16)(x1 * s + x2 * c);
}

// V repack+transpose: Vf (B*T, KV*HD) fp32 -> Vt (B, KV, HD, T) f16
__global__ void v_pack(const float* __restrict__ Vf, _Float16* __restrict__ Vt) {
  int idx = blockIdx.x * blockDim.x + threadIdx.x;
  int total = B_SZ * N_KV * HD * T_SZ;
  if (idx >= total) return;
  int tt = idx % T_SZ;
  int d  = (idx / T_SZ) % HD;
  int kh = (idx / (T_SZ * HD)) % N_KV;
  int b  = idx / (T_SZ * HD * N_KV);
  Vt[idx] = (_Float16)Vf[(size_t)(b * T_SZ + tt) * (N_KV * HD) + kh * HD + d];
}

// ---------------------------------------------------------------------------
// Flash attention, 1 wave per 16 query rows of one (b, h).
// S = Q K^T via WMMA (K-tile of 32 keys = 2 n-tiles x 4 k-steps),
// online softmax with ALiBi + causal, O += P V via WMMA (P staged through LDS
// to convert C-layout -> A-layout). Single-wave workgroup so __syncthreads is
// safe despite causality-dependent trip counts.
// ---------------------------------------------------------------------------
__global__ __launch_bounds__(32)
void flash_attn(const _Float16* __restrict__ Qh, const _Float16* __restrict__ Kh,
                const _Float16* __restrict__ Vt, _Float16* __restrict__ Oh) {
  __shared__ __align__(16) _Float16 Ps[16 * 32];

  const int lane = threadIdx.x;
  const int qb = blockIdx.x;          // T/16 q-blocks
  const int bh = blockIdx.y;          // B*H
  const int b  = bh >> 4;
  const int h  = bh & 15;
  const int kh = h >> 2;              // groups = H / KV = 4 (jnp.repeat semantics)
  const int q0 = qb * 16;
  const int r  = lane & 15;
  const int hs = lane >> 4;

  const _Float16* Qb = Qh + (size_t)bh * T_SZ * HD;
  const _Float16* Kb = Kh + (size_t)(b * N_KV + kh) * T_SZ * HD;
  const _Float16* Vb = Vt + (size_t)(b * N_KV + kh) * HD * T_SZ;

  const float slope = exp2f(-8.0f * (float)(h + 1) / 16.0f);
  const float scale = 0.08838834764831845f;   // 1/sqrt(128)
  const float NEG_INF = -__builtin_inff();

  // Q fragments for this 16-row block (K = 128 -> 4 WMMA k-steps), loaded once
  v16h qf[4];
#pragma unroll
  for (int s = 0; s < 4; s++) qf[s] = ldfragA(Qb + (size_t)q0 * HD + s * 32, HD, lane);

  v8f o[8];
#pragma unroll
  for (int dt = 0; dt < 8; dt++) o[dt] = 0.0f;
  float mi[8], li[8];
#pragma unroll
  for (int i = 0; i < 8; i++) { mi[i] = NEG_INF; li[i] = 0.0f; }

  const int nkt = (q0 + 16 + 31) >> 5;   // key tiles of 32, causal bound
  for (int kt = 0; kt < nkt; kt++) {
    const int k0 = kt * 32;

    v8f Sv[2];
    Sv[0] = 0.0f; Sv[1] = 0.0f;
#pragma unroll
    for (int nt = 0; nt < 2; nt++)
#pragma unroll
      for (int s = 0; s < 4; s++) {
        v16h kf = ldfragB(Kb + (size_t)(k0 + nt * 16) * HD + s * 32, HD, lane);
        Sv[nt] = wmma32(qf[s], kf, Sv[nt]);
      }

    const int key0 = k0 + r;
    const int key1 = k0 + 16 + r;
    float p0[8], p1[8];
#pragma unroll
    for (int i = 0; i < 8; i++) {
      int row = q0 + hs * 8 + i;
      float v0 = (key0 <= row) ? (Sv[0][i] * scale + slope * (float)(key0 - row)) : NEG_INF;
      float v1 = (key1 <= row) ? (Sv[1][i] * scale + slope * (float)(key1 - row)) : NEG_INF;
      float tm = fmaxf(v0, v1);
      tm = fmaxf(tm, __shfl_xor(tm, 1));
      tm = fmaxf(tm, __shfl_xor(tm, 2));
      tm = fmaxf(tm, __shfl_xor(tm, 4));
      tm = fmaxf(tm, __shfl_xor(tm, 8));
      float nm   = fmaxf(mi[i], tm);             // finite after tile 0 for every row
      float corr = __expf(mi[i] - nm);
      float e0 = __expf(v0 - nm);
      float e1 = __expf(v1 - nm);
      float rs = e0 + e1;
      rs += __shfl_xor(rs, 1);
      rs += __shfl_xor(rs, 2);
      rs += __shfl_xor(rs, 4);
      rs += __shfl_xor(rs, 8);
      li[i] = li[i] * corr + rs;
      mi[i] = nm;
      p0[i] = e0; p1[i] = e1;
#pragma unroll
      for (int dt = 0; dt < 8; dt++) o[dt][i] *= corr;
    }

    // stage P (C-layout) into LDS row-major 16x32, re-read as A-layout fragment
    __syncthreads();
#pragma unroll
    for (int i = 0; i < 8; i++) {
      Ps[(hs * 8 + i) * 32 + r]      = (_Float16)p0[i];
      Ps[(hs * 8 + i) * 32 + 16 + r] = (_Float16)p1[i];
    }
    __syncthreads();
    v16h pf = ldfragA(Ps, 32, lane);

#pragma unroll
    for (int dt = 0; dt < 8; dt++) {
      v16h vf = ldfragB(Vb + (size_t)(dt * 16) * T_SZ + k0, T_SZ, lane);
      o[dt] = wmma32(pf, vf, o[dt]);
    }
  }

  // normalize and write attn output f16 in (B, T, H*HD) layout for final GEMM
#pragma unroll
  for (int i = 0; i < 8; i++) {
    int trow = q0 + hs * 8 + i;
    float inv = 1.0f / li[i];
    _Float16* orow = Oh + (size_t)(b * T_SZ + trow) * D_MODEL + h * HD;
#pragma unroll
    for (int dt = 0; dt < 8; dt++) orow[dt * 16 + r] = (_Float16)(o[dt][i] * inv);
  }
}

// ---------------------------------------------------------------------------
extern "C" void kernel_launch(void* const* d_in, const int* in_sizes, int n_in,
                              void* d_out, int out_size, void* d_ws, size_t ws_size,
                              hipStream_t stream) {
  const float* x  = (const float*)d_in[0];
  const float* Wq = (const float*)d_in[1];
  const float* Wk = (const float*)d_in[2];
  const float* Wv = (const float*)d_in[3];
  const float* Wo = (const float*)d_in[4];
  float* out = (float*)d_out;

  char* w = (char*)d_ws;
  size_t off = 0;
  auto carve = [&](size_t bytes) -> void* {
    void* p = (void*)(w + off);
    off += (bytes + 255) & ~(size_t)255;
    return p;
  };

  _Float16* xh  = (_Float16*)carve((size_t)M_ROWS * D_MODEL * 2);
  _Float16* Wqt = (_Float16*)carve((size_t)D_MODEL * D_MODEL * 2);
  _Float16* Wkt = (_Float16*)carve((size_t)512 * D_MODEL * 2);
  _Float16* Wvt = (_Float16*)carve((size_t)512 * D_MODEL * 2);
  _Float16* Wot = (_Float16*)carve((size_t)D_MODEL * D_MODEL * 2);
  float*    Qf  = (float*)carve((size_t)M_ROWS * D_MODEL * 4);
  float*    Kf  = (float*)carve((size_t)M_ROWS * 512 * 4);
  float*    Vf  = (float*)carve((size_t)M_ROWS * 512 * 4);
  _Float16* Qh  = (_Float16*)carve((size_t)B_SZ * N_HEADS * T_SZ * HD * 2);
  _Float16* Kh  = (_Float16*)carve((size_t)B_SZ * N_KV * T_SZ * HD * 2);
  _Float16* Vt  = (_Float16*)carve((size_t)B_SZ * N_KV * HD * T_SZ * 2);
  _Float16* Ah  = (_Float16*)carve((size_t)M_ROWS * D_MODEL * 2);

  int n;

  n = M_ROWS * D_MODEL;
  f32_to_f16<<<(n + 255) / 256, 256, 0, stream>>>(x, xh, n);

  n = D_MODEL * D_MODEL;
  transpose_to_f16<<<(n + 255) / 256, 256, 0, stream>>>(Wq, Wqt, D_MODEL, D_MODEL);
  n = D_MODEL * 512;
  transpose_to_f16<<<(n + 255) / 256, 256, 0, stream>>>(Wk, Wkt, D_MODEL, 512);
  transpose_to_f16<<<(n + 255) / 256, 256, 0, stream>>>(Wv, Wvt, D_MODEL, 512);
  n = D_MODEL * D_MODEL;
  transpose_to_f16<<<(n + 255) / 256, 256, 0, stream>>>(Wo, Wot, D_MODEL, D_MODEL);

  gemm_f16t<<<dim3(D_MODEL / 128, M_ROWS / 128), 256, 0, stream>>>(xh, Wqt, Qf, M_ROWS, D_MODEL, D_MODEL);
  gemm_f16t<<<dim3(512 / 128,     M_ROWS / 128), 256, 0, stream>>>(xh, Wkt, Kf, M_ROWS, 512, D_MODEL);
  gemm_f16t<<<dim3(512 / 128,     M_ROWS / 128), 256, 0, stream>>>(xh, Wvt, Vf, M_ROWS, 512, D_MODEL);

  n = B_SZ * N_HEADS * T_SZ * 64;
  rope_pack<<<(n + 255) / 256, 256, 0, stream>>>(Qf, Qh, N_HEADS);
  n = B_SZ * N_KV * T_SZ * 64;
  rope_pack<<<(n + 255) / 256, 256, 0, stream>>>(Kf, Kh, N_KV);
  n = B_SZ * N_KV * HD * T_SZ;
  v_pack<<<(n + 255) / 256, 256, 0, stream>>>(Vf, Vt);

  flash_attn<<<dim3(T_SZ / 16, B_SZ * N_HEADS), 32, 0, stream>>>(Qh, Kh, Vt, Ah);

  gemm_f16t<<<dim3(D_MODEL / 128, M_ROWS / 128), 256, 0, stream>>>(Ah, Wot, out, M_ROWS, D_MODEL, D_MODEL);
}
